// DyGCN_58643483459899
// MI455X (gfx1250) — matrix operand order repaired and checked by reading, not compiled, not run on hardware
//
#include <hip/hip_runtime.h>
#include <math.h>

// Problem constants (match reference)
constexpr int kB = 8, kN = 1024, kC = 64, kT = 12, kO = 64;
constexpr int kTB = kT * kB;

typedef _Float16 v16h __attribute__((ext_vector_type(16)));
typedef _Float16 h2   __attribute__((ext_vector_type(2)));
typedef float    v8f  __attribute__((ext_vector_type(8)));

// Hardware tanh (CDNA5 V_TANH_F32). Builtin when available (compiler manages the
// TRANS-op hazard); otherwise raw ISA with a v_nop to satisfy the 1-op gap rule.
static __device__ __forceinline__ float fast_tanh(float x) {
#if __has_builtin(__builtin_amdgcn_tanhf)
  return __builtin_amdgcn_tanhf(x);
#else
  float r;
  asm("v_tanh_f32 %0, %1\n\tv_nop" : "=v"(r) : "v"(x));
  return r;
#endif
}

static __device__ __forceinline__ v8f wmma_f16(v16h a, v16h b, v8f c) {
  // D = A(16x32 f16) * B(32x16 f16) + C(16x16 f32)
  return __builtin_amdgcn_wmma_f32_16x16x32_f16(false, a, false, b, (short)0, c,
                                                false, false);
}

// A-fragment (16x32, f16): lane l holds row (l&15); K mapping per CDNA5 ISA 7.12.2:
// lanes 0-15: VGPR j<4 -> K=2j..2j+1, j>=4 -> K=16+2(j-4); lanes 16-31: +8.
static __device__ __forceinline__ v16h load_a_frag(const _Float16* __restrict__ src,
                                                   int stride) {
  int l = threadIdx.x & 31;
  const _Float16* p = src + (size_t)(l & 15) * stride + ((l >> 4) << 3);
  v16h a;
#pragma unroll
  for (int s = 0; s < 8; ++s) {
    int rel = (s < 4) ? (2 * s) : (16 + 2 * (s - 4));
    h2 v = *(const h2*)(p + rel);
    a[2 * s] = v[0];
    a[2 * s + 1] = v[1];
  }
  return a;
}

// B-fragment (32x16, f16): lane l holds column (l&15); VGPR j<4 -> K=2j..2j+1, j>=4 -> K=16+2(j-4); lanes 16-31: +8.
// src is "column-major": element B[k][col] = src[col*stride + k]  (K contiguous).
static __device__ __forceinline__ v16h load_b_frag(const _Float16* __restrict__ src,
                                                   int stride) {
  int l = threadIdx.x & 31;
  const _Float16* p = src + (size_t)(l & 15) * stride + ((l >> 4) << 3);
  v16h b;
#pragma unroll
  for (int s = 0; s < 8; ++s) {
    h2 v = *(const h2*)(p + ((s < 4) ? (2 * s) : (16 + 2 * (s - 4))));
    b[2 * s] = v[0];
    b[2 * s + 1] = v[1];
  }
  return b;
}

// ---------------- Kernel 1: tiny conversions (adjacency f16, Wd transposed f16) ----
__global__ __launch_bounds__(256) void prep_small_k(const float* __restrict__ adj,
                                                    const float* __restrict__ Wd,
                                                    _Float16* __restrict__ adjh,
                                                    _Float16* __restrict__ WdhT) {
  int idx = blockIdx.x * 256 + threadIdx.x;
  if (idx < kN * kN) {
    adjh[idx] = (_Float16)adj[idx];
    return;
  }
  idx -= kN * kN;
  if (idx < kT * kO * kC) {
    int t = idx / (kO * kC);
    int r = idx % (kO * kC);
    int o = r / kC;
    int c = r % kC;
    // WdhT[t][o][c] = Wd[t][c][o]  (so K=c is contiguous for B-fragments)
    WdhT[idx] = (_Float16)Wd[((size_t)t * kC + c) * kO + o];
  }
}

// ---------------- Kernel 2: per-(t,b,n) linears, tanh gates, layout changes -------
// Produces: DE1h/DE2h [tb][n][o] f16 (row-major, K=o contiguous),
//           xthT [tb][c][n] f16, Yh=x@Ws [tb][o][n] f16 (K=n contiguous for B-frags),
//           out initialized with 0.05*x0 + 0.475*(bs+bd).
__global__ __launch_bounds__(256) void prep_main_k(
    const float* __restrict__ x, const float* __restrict__ W0,
    const float* __restrict__ b0, const float* __restrict__ Ws,
    const float* __restrict__ bs, const float* __restrict__ Wd,
    const float* __restrict__ bd, const float* __restrict__ E1,
    const float* __restrict__ E2, _Float16* __restrict__ DE1h,
    _Float16* __restrict__ DE2h, _Float16* __restrict__ xthT,
    _Float16* __restrict__ Yh, float* __restrict__ out) {
  __shared__ float xs[4][kC];
  int lr = threadIdx.x >> 6;  // local row 0..3
  int o = threadIdx.x & 63;   // doubles as channel index c
  int blk = blockIdx.x;       // over kTB * (kN/4)
  int nQuad = blk % (kN / 4);
  int tb = blk / (kN / 4);
  int t = tb / kB, b = tb % kB;
  int n = nQuad * 4 + lr;

  // x[b,n,c,t]  (stride kT over c)
  xs[lr][o] = x[(((size_t)b * kN + n) * kC + o) * kT + t];
  __syncthreads();

  float accW0 = b0[o];
  float accWd = bd[t * kO + o];
  float accWs = 0.f;
#pragma unroll 8
  for (int c = 0; c < kC; ++c) {
    float xc = xs[lr][c];
    accW0 = fmaf(xc, W0[c * kO + o], accW0);
    accWd = fmaf(xc, Wd[((size_t)t * kC + c) * kO + o], accWd);
    accWs = fmaf(xc, Ws[((size_t)t * kC + c) * kO + o], accWs);
  }
  size_t tbn = (size_t)tb * kN + n;
  float e1 = E1[((size_t)t * kN + n) * kO + o];
  float e2 = E2[((size_t)t * kN + n) * kO + o];
  DE1h[tbn * kO + o] = (_Float16)fast_tanh(accWd * e1);  // ALPHA_PRE = 1
  DE2h[tbn * kO + o] = (_Float16)fast_tanh(accWd * e2);
  Yh[((size_t)tb * kO + o) * kN + n] = (_Float16)accWs;
  xthT[((size_t)tb * kC + o) * kN + n] = (_Float16)xs[lr][o];
  // fused = 0.475*static + 0.475*dyn + 0.05*x0 ; biases folded in here
  out[(((size_t)b * kN + n) * kO + o) * kT + t] =
      0.05f * accW0 + 0.475f * (bs[t * kO + o] + bd[t * kO + o]);
}

// ---------------- Kernel 3: row sums of DA -> d^{-1/2}  ---------------------------
// One wave per (tb, 16-row block). A[n,m] = relu(tanh(M[n,m]-M[m,n])), diag adds 1.
__global__ __launch_bounds__(256) void rowsum_k(const _Float16* __restrict__ DE1h,
                                                const _Float16* __restrict__ DE2h,
                                                float* __restrict__ dinv) {
  int wave = blockIdx.x * 8 + (threadIdx.x >> 5);
  int nb = wave & 63;
  int tb = wave >> 6;
  int n0 = nb << 4;
  int l = threadIdx.x & 31, hi = l >> 4;

  const _Float16* d1 = DE1h + (size_t)tb * kN * kO;
  const _Float16* d2 = DE2h + (size_t)tb * kN * kO;

  v16h aP0 = load_a_frag(d1 + (size_t)n0 * kO, kO);
  v16h aP1 = load_a_frag(d1 + (size_t)n0 * kO + 32, kO);
  v16h aQ0 = load_a_frag(d2 + (size_t)n0 * kO, kO);
  v16h aQ1 = load_a_frag(d2 + (size_t)n0 * kO + 32, kO);

  float racc[8] = {0.f, 0.f, 0.f, 0.f, 0.f, 0.f, 0.f, 0.f};
  for (int m0 = 0; m0 < kN; m0 += 16) {
    v16h bP0 = load_b_frag(d2 + (size_t)m0 * kO, kO);
    v16h bP1 = load_b_frag(d2 + (size_t)m0 * kO + 32, kO);
    v16h bQ0 = load_b_frag(d1 + (size_t)m0 * kO, kO);
    v16h bQ1 = load_b_frag(d1 + (size_t)m0 * kO + 32, kO);
    v8f P = {};
    v8f Q = {};
    P = wmma_f16(aP0, bP0, P);
    P = wmma_f16(aP1, bP1, P);
    Q = wmma_f16(aQ0, bQ0, Q);  // computes M^T[n,m] directly
    Q = wmma_f16(aQ1, bQ1, Q);
#pragma unroll
    for (int i = 0; i < 8; ++i) {
      float v = fast_tanh(P[i] - Q[i]);
      racc[i] += (v > 0.f) ? v : 0.f;
    }
  }
#pragma unroll
  for (int i = 0; i < 8; ++i) {
    float r = racc[i];
    r += __shfl_xor(r, 1);
    r += __shfl_xor(r, 2);
    r += __shfl_xor(r, 4);
    r += __shfl_xor(r, 8);
    if ((l & 15) == 0)
      dinv[(size_t)tb * kN + n0 + i + (hi << 3)] =
          rsqrtf(fmaxf(1.0f + r, 1e-12f));
  }
}

// ---------------- Kernel 4: fused dynamic + static GEMMs + output -----------------
__global__ __launch_bounds__(256) void final_k(
    const _Float16* __restrict__ DE1h, const _Float16* __restrict__ DE2h,
    const _Float16* __restrict__ xthT, const _Float16* __restrict__ Yh,
    const _Float16* __restrict__ adjh, const _Float16* __restrict__ WdhT,
    const float* __restrict__ dinv, float* __restrict__ out) {
  __shared__ _Float16 stbuf[8 * 16 * 64];  // per-wave private staging (1024 f16 each)
  int wid = threadIdx.x >> 5;
  _Float16* st = stbuf + wid * (16 * 64);

  int wave = blockIdx.x * 8 + wid;
  int nb = wave & 63;
  int tb = wave >> 6;
  int n0 = nb << 4;
  int t = tb / kB, b = tb % kB;
  int l = threadIdx.x & 31, hi = l >> 4, colr = l & 15;

  const _Float16* d1 = DE1h + (size_t)tb * kN * kO;
  const _Float16* d2 = DE2h + (size_t)tb * kN * kO;
  const _Float16* xT = xthT + (size_t)tb * kC * kN;
  const _Float16* yT = Yh + (size_t)tb * kO * kN;

  v16h aP0 = load_a_frag(d1 + (size_t)n0 * kO, kO);
  v16h aP1 = load_a_frag(d1 + (size_t)n0 * kO + 32, kO);
  v16h aQ0 = load_a_frag(d2 + (size_t)n0 * kO, kO);
  v16h aQ1 = load_a_frag(d2 + (size_t)n0 * kO + 32, kO);

  float dvn[8];
#pragma unroll
  for (int i = 0; i < 8; ++i)
    dvn[i] = dinv[(size_t)tb * kN + n0 + i + (hi << 3)];

  v8f dacc[4] = {{}, {}, {}, {}};
  v8f sacc[4] = {{}, {}, {}, {}};

  for (int mc = 0; mc < kN; mc += 32) {
    // Build one 16x32 DA chunk (normalized, +I) into per-wave LDS.
#pragma unroll
    for (int half = 0; half < 2; ++half) {
      int m0 = mc + (half << 4);
      v16h bP0 = load_b_frag(d2 + (size_t)m0 * kO, kO);
      v16h bP1 = load_b_frag(d2 + (size_t)m0 * kO + 32, kO);
      v16h bQ0 = load_b_frag(d1 + (size_t)m0 * kO, kO);
      v16h bQ1 = load_b_frag(d1 + (size_t)m0 * kO + 32, kO);
      v8f P = {};
      v8f Q = {};
      P = wmma_f16(aP0, bP0, P);
      P = wmma_f16(aP1, bP1, P);
      Q = wmma_f16(aQ0, bQ0, Q);
      Q = wmma_f16(aQ1, bQ1, Q);
      float dvm = dinv[(size_t)tb * kN + m0 + colr];
#pragma unroll
      for (int i = 0; i < 8; ++i) {
        float v = fast_tanh(P[i] - Q[i]);
        v = (v > 0.f) ? v : 0.f;
        if (n0 == m0 && (i + (hi << 3)) == colr) v += 1.0f;  // +I
        v *= dvn[i] * dvm;                                   // D^-1/2 (A+I) D^-1/2
        st[(i + (hi << 3)) * 32 + (half << 4) + colr] = (_Float16)v;
      }
    }
    // Intra-wave LDS store->load ordering (private region, no block barrier needed).
    asm volatile("s_wait_dscnt 0" ::: "memory");

    v16h aDA = load_a_frag(st, 32);
    v16h aAdj = load_a_frag(adjh + (size_t)n0 * kN + mc, kN);
    if (mc + 32 < kN)
      __builtin_prefetch(adjh + (size_t)n0 * kN + mc + 32, 0, 1);

#pragma unroll
    for (int ct = 0; ct < 4; ++ct) {  // dagg[n,c] += DA * x
      v16h bX = load_b_frag(xT + (size_t)(ct << 4) * kN + mc, kN);
      dacc[ct] = wmma_f16(aDA, bX, dacc[ct]);
    }
#pragma unroll
    for (int ot = 0; ot < 4; ++ot) {  // static[n,o] += adj * (x@Ws)
      v16h bY = load_b_frag(yT + (size_t)(ot << 4) * kN + mc, kN);
      sacc[ot] = wmma_f16(aAdj, bY, sacc[ot]);
    }
    asm volatile("s_wait_dscnt 0" ::: "memory");
  }

  // Epilogue: dyn_out = dagg @ Wd  (stage dagg as f16 A-fragments via LDS)
#pragma unroll
  for (int ct = 0; ct < 4; ++ct)
#pragma unroll
    for (int i = 0; i < 8; ++i)
      st[(i + (hi << 3)) * 64 + (ct << 4) + colr] = (_Float16)dacc[ct][i];
  asm volatile("s_wait_dscnt 0" ::: "memory");

  const _Float16* wdt = WdhT + (size_t)t * kO * kC;
#pragma unroll
  for (int ot = 0; ot < 4; ++ot) {
    v8f dy = {};
#pragma unroll
    for (int kk = 0; kk < 2; ++kk) {
      v16h aD = load_a_frag(st + kk * 32, 64);
      v16h bW = load_b_frag(wdt + (size_t)(ot << 4) * kC + kk * 32, kC);
      dy = wmma_f16(aD, bW, dy);
    }
#pragma unroll
    for (int i = 0; i < 8; ++i) {
      int n = n0 + i + (hi << 3);
      int o = (ot << 4) + colr;
      size_t idx = (((size_t)b * kN + n) * kO + o) * kT + t;
      out[idx] += 0.475f * (sacc[ot][i] + dy[i]);  // biases already folded in prep
    }
  }
}

// ---------------- Host launcher ---------------------------------------------------
extern "C" void kernel_launch(void* const* d_in, const int* in_sizes, int n_in,
                              void* d_out, int out_size, void* d_ws, size_t ws_size,
                              hipStream_t stream) {
  (void)in_sizes; (void)n_in; (void)out_size; (void)ws_size;
  const float* x   = (const float*)d_in[0];
  const float* adj = (const float*)d_in[1];
  const float* W0  = (const float*)d_in[2];
  const float* b0  = (const float*)d_in[3];
  const float* Ws  = (const float*)d_in[4];
  const float* bs  = (const float*)d_in[5];
  const float* Wd  = (const float*)d_in[6];
  const float* bd  = (const float*)d_in[7];
  const float* E1  = (const float*)d_in[8];
  const float* E2  = (const float*)d_in[9];
  float* out = (float*)d_out;

  char* ws = (char*)d_ws;
  size_t off = 0;
  auto suballoc = [&](size_t bytes) -> void* {
    void* p = ws + off;
    off = (off + bytes + 255) & ~(size_t)255;
    return p;
  };
  size_t nDE = (size_t)kTB * kN * kO;
  _Float16* DE1h = (_Float16*)suballoc(nDE * 2);
  _Float16* DE2h = (_Float16*)suballoc(nDE * 2);
  _Float16* xthT = (_Float16*)suballoc((size_t)kTB * kC * kN * 2);
  _Float16* Yh   = (_Float16*)suballoc((size_t)kTB * kO * kN * 2);
  _Float16* adjh = (_Float16*)suballoc((size_t)kN * kN * 2);
  _Float16* WdhT = (_Float16*)suballoc((size_t)kT * kO * kC * 2);
  float*    dinv = (float*)suballoc((size_t)kTB * kN * 4);

  int smallTotal = kN * kN + kT * kO * kC;
  prep_small_k<<<(smallTotal + 255) / 256, 256, 0, stream>>>(adj, Wd, adjh, WdhT);
  prep_main_k<<<kTB * (kN / 4), 256, 0, stream>>>(x, W0, b0, Ws, bs, Wd, bd, E1, E2,
                                                  DE1h, DE2h, xthT, Yh, out);
  int nWaves = kTB * (kN / 16);  // 6144
  rowsum_k<<<nWaves / 8, 256, 0, stream>>>(DE1h, DE2h, dinv);
  final_k<<<nWaves / 8, 256, 0, stream>>>(DE1h, DE2h, xthT, Yh, adjh, WdhT, dinv, out);
}
